// QuantizedLinearLayerWithActivation_6476810682793
// MI455X (gfx1250) — matrix operand order repaired and checked by reading, not compiled, not run on hardware
//
#include <hip/hip_runtime.h>

typedef __attribute__((ext_vector_type(16))) _Float16 v16h;
typedef __attribute__((ext_vector_type(8)))  _Float16 v8h;
typedef __attribute__((ext_vector_type(8)))  float    v8f;

#define BM 128
#define BN 128
#define BK 32
#define LDS_STRIDE 40   // 32 halves + 8 pad (80B row pitch -> spreads LDS banks)

__global__ __launch_bounds__(256)
void qlinear_wmma_f16_kernel(const float* __restrict__ x,       // [M,K] fp32
                             const float* __restrict__ weight,  // [N,K] fp32 (int8-valued)
                             const float* __restrict__ scale,   // [N]
                             const float* __restrict__ bias,    // [N]
                             const float* __restrict__ act_scale_p,
                             float* __restrict__ out,           // [M,N]
                             int M, int N, int K)
{
    __shared__ _Float16 sA[BM * LDS_STRIDE];
    __shared__ _Float16 sB[BN * LDS_STRIDE];

    const int tid  = threadIdx.x;
    const int lane = tid & 31;
    const int wave = tid >> 5;
    const int wm   = wave >> 2;   // 0..1 : 64-row slab
    const int wn   = wave & 3;    // 0..3 : 32-col slab

    const int bm0 = blockIdx.y * BM;
    const int bn0 = blockIdx.x * BN;

    const float act     = act_scale_p[0];
    const float inv_act = 1.0f / act;

    // Global staging: 8 threads cover one 32-float row (8 x float4); 32 rows/pass, 4 passes.
    const int lrow8 = tid >> 3;        // 0..31
    const int lcol4 = (tid & 7) * 4;   // 0,4,...,28

    float4 rA[4], rB[4];

    auto loadTiles = [&](int k0) {
        #pragma unroll
        for (int p = 0; p < 4; ++p) {
            const int row = p * 32 + lrow8;
            rA[p] = *(const float4*)(x      + (size_t)(bm0 + row) * K + k0 + lcol4);
            rB[p] = *(const float4*)(weight + (size_t)(bn0 + row) * K + k0 + lcol4);
        }
    };

    auto quant = [&](float v) -> _Float16 {
        float q = rintf(v * inv_act);                     // round-half-even, like jnp.round
        q = fminf(fmaxf(q, -32768.0f), 32767.0f);
        return (_Float16)q;                               // exact: |q| << 2048 in practice
    };

    auto stashTiles = [&]() {
        #pragma unroll
        for (int p = 0; p < 4; ++p) {
            const int row = p * 32 + lrow8;
            _Float16* a = &sA[row * LDS_STRIDE + lcol4];
            a[0] = quant(rA[p].x); a[1] = quant(rA[p].y);
            a[2] = quant(rA[p].z); a[3] = quant(rA[p].w);
            _Float16* b = &sB[row * LDS_STRIDE + lcol4];
            b[0] = (_Float16)rB[p].x; b[1] = (_Float16)rB[p].y;
            b[2] = (_Float16)rB[p].z; b[3] = (_Float16)rB[p].w;
        }
    };

    v8f acc[4][2];
    #pragma unroll
    for (int i = 0; i < 4; ++i)
        #pragma unroll
        for (int j = 0; j < 2; ++j)
            acc[i][j] = {};

    loadTiles(0);

    // Fragment addressing per CDNA5 WMMA VGPR layouts:
    //  A (16x32 f16): lane&15 = M row; lanes 0-15 hold K 0-7 & 16-23, lanes 16-31 hold K 8-15 & 24-31
    //  B (32x16 f16): lane&15 = N col; lanes 0-15 hold K 0-7 & 16-23, lanes 16-31 hold K 8-15 & 24-31
    const int lrow = lane & 15;
    const int khiA = (lane >> 4) * 8;
    const int khiB = (lane >> 4) * 8;

    for (int k0 = 0; k0 < K; k0 += BK) {
        stashTiles();
        __syncthreads();
        if (k0 + BK < K) loadTiles(k0 + BK);   // prefetch next tile -> overlaps WMMA below

        v16h afrag[4];
        #pragma unroll
        for (int im = 0; im < 4; ++im) {
            const _Float16* p = &sA[(wm * 64 + im * 16 + lrow) * LDS_STRIDE + khiA];
            v8h lo = *(const v8h*)p;           // K khiA..khiA+7   (ds_load_b128)
            v8h hi = *(const v8h*)(p + 16);    // K khiA+16..+23
            afrag[im] = __builtin_shufflevector(lo, hi,
                0,1,2,3,4,5,6,7,8,9,10,11,12,13,14,15);
        }
        v16h bfrag[2];
        #pragma unroll
        for (int jn = 0; jn < 2; ++jn) {
            const _Float16* p = &sB[(wn * 32 + jn * 16 + lrow) * LDS_STRIDE + khiB];
            v8h lo = *(const v8h*)p;           // K khiB..khiB+7
            v8h hi = *(const v8h*)(p + 16);    // K khiB+16..+23
            bfrag[jn] = __builtin_shufflevector(lo, hi,
                0,1,2,3,4,5,6,7,8,9,10,11,12,13,14,15);
        }

        #pragma unroll
        for (int im = 0; im < 4; ++im)
            #pragma unroll
            for (int jn = 0; jn < 2; ++jn)
                acc[im][jn] = __builtin_amdgcn_wmma_f32_16x16x32_f16(
                    false, afrag[im], false, bfrag[jn],
                    (short)0, acc[im][jn], false, false);

        __syncthreads();
    }

    // Epilogue: C/D f32 layout -> lane&15 = N col, VGPR r = M row r (+8 for lanes 16-31)
    const int rbase = (lane >> 4) * 8;
    #pragma unroll
    for (int jn = 0; jn < 2; ++jn) {
        const int n  = bn0 + wn * 32 + jn * 16 + lrow;
        const float s  = act * scale[n];
        const float bv = bias[n];
        #pragma unroll
        for (int im = 0; im < 4; ++im) {
            const int m0 = bm0 + wm * 64 + im * 16 + rbase;
            #pragma unroll
            for (int r = 0; r < 8; ++r)
                out[(size_t)(m0 + r) * N + n] = acc[im][jn][r] * s + bv;
        }
    }
}

extern "C" void kernel_launch(void* const* d_in, const int* in_sizes, int n_in,
                              void* d_out, int out_size, void* d_ws, size_t ws_size,
                              hipStream_t stream) {
    const float* x      = (const float*)d_in[0];
    const float* weight = (const float*)d_in[1];
    const float* scale  = (const float*)d_in[2];
    const float* bias   = (const float*)d_in[3];
    const float* actsc  = (const float*)d_in[4];
    float* out = (float*)d_out;

    const int N = in_sizes[2];           // 4096 (scale length = D_OUT)
    const int K = in_sizes[1] / N;       // 4096
    const int M = in_sizes[0] / K;       // 8192 (B*S)

    dim3 grid(N / BN, M / BM);           // 32 x 64
    dim3 block(256);
    qlinear_wmma_f16_kernel<<<grid, block, 0, stream>>>(
        x, weight, scale, bias, actsc, out, M, N, K);
}
